// GL_TAGConv_3l_512h_wnw_k3_44753559224339
// MI455X (gfx1250) — matrix operand order repaired and checked by reading, not compiled, not run on hardware
//
#include <hip/hip_runtime.h>
#include <hip/hip_bf16.h>
#include <math.h>

// ---------------------------------------------------------------------------
// GL_TAGConv for MI455X. Per layer: SpMM chain (h1,h2,h3), then ONE fused
// WMMA GEMM: out = elu( [x|h1|h2|h3] @ Wflat + b ), Kd = 4*fin, bf16 tiles,
// f32 accumulate, double-buffered LDS, v_wmma_f32_16x16x32_bf16.
// GEMM hot loop: no lambdas/arrays/struct-vectors -> SROA-friendly, no spills.
// ---------------------------------------------------------------------------

typedef __attribute__((ext_vector_type(16))) __bf16 v16bf;
typedef __attribute__((ext_vector_type(8)))  __bf16 v8bf;
typedef __attribute__((ext_vector_type(4)))  __bf16 v4bf;
typedef __attribute__((ext_vector_type(4)))  float  v4f;
typedef __attribute__((ext_vector_type(8)))  float  v8f;

__device__ __forceinline__ v16bf cat16(v8bf lo, v8bf hi8) {
    return __builtin_shufflevector(lo, hi8,
        0,1,2,3,4,5,6,7,8,9,10,11,12,13,14,15);
}

// ------------------------------ utility ------------------------------------

__global__ void zero_kernel(float* __restrict__ p, size_t n) {
    size_t i = (size_t)blockIdx.x * blockDim.x + threadIdx.x;
    if (i < n) p[i] = 0.0f;
}

__global__ void deg_kernel(const int* __restrict__ dst, const float* __restrict__ w,
                           float* __restrict__ deg_w, float* __restrict__ deg_n, int E) {
    int e = blockIdx.x * blockDim.x + threadIdx.x;
    if (e < E) {
        int d = dst[e];
        atomicAdd(&deg_w[d], w[e]);
        atomicAdd(&deg_n[d], 1.0f);
    }
}

__global__ void norm_kernel(const int* __restrict__ src, const int* __restrict__ dst,
                            const float* __restrict__ w,
                            const float* __restrict__ deg_w, const float* __restrict__ deg_n,
                            float* __restrict__ norm_w, float* __restrict__ norm_n, int E) {
    int e = blockIdx.x * blockDim.x + threadIdx.x;
    if (e < E) {
        int s = src[e], d = dst[e];
        float dws = deg_w[s], dwd = deg_w[d];
        float iws = dws > 0.f ? rsqrtf(dws) : 0.f;
        float iwd = dwd > 0.f ? rsqrtf(dwd) : 0.f;
        norm_w[e] = iws * w[e] * iwd;
        float dns = deg_n[s], dnd = deg_n[d];
        float ins = dns > 0.f ? rsqrtf(dns) : 0.f;
        float ind = dnd > 0.f ? rsqrtf(dnd) : 0.f;
        norm_n[e] = ins * ind;
    }
}

// SpMM scatter: hout[dst] += norm[e] * hin[src]. One thread = 4 features of one edge.
__global__ void spmm_kernel(const int* __restrict__ src, const int* __restrict__ dst,
                            const float* __restrict__ norm,
                            const float* __restrict__ hin, float* __restrict__ hout,
                            int E, int F4) {
    long tid = (long)blockIdx.x * blockDim.x + threadIdx.x;
    long total = (long)E * F4;
    if (tid >= total) return;
    int e = (int)(tid / F4);
    int f = (int)(tid % F4) * 4;
    int Fdim = F4 * 4;
    const v4f v = *(const v4f*)&hin[(size_t)src[e] * Fdim + f];
    float nw = norm[e];
    float* o = &hout[(size_t)dst[e] * Fdim + f];
    atomicAdd(o + 0, v[0] * nw);
    atomicAdd(o + 1, v[1] * nw);
    atomicAdd(o + 2, v[2] * nw);
    atomicAdd(o + 3, v[3] * nw);
}

// ------------------------------ fused WMMA GEMM ----------------------------
// C[M x Nd] = elu( [A0|A1|A2|A3] @ B + bias ), A_j are M x fin, B is 4fin x Nd.
// Block: 256 threads (8 waves). Tile 128x64; wave owns 32x32 (2x2 WMMAs).

#define TM 128
#define TN 64
#define TK 32
#define ASTR 40   // LDS row stride (bf16): 80B rows keep b128 frags 16B-aligned
#define BSTR 40

__device__ __forceinline__ void store_strip(float* cp, int mBase, int M, int Nd,
                                            v8f a0, v8f a1, float b0, float b1) {
    if (mBase + 8 <= M) {
        #pragma unroll
        for (int r = 0; r < 8; ++r) {
            float v0 = a0[r] + b0;
            float v1 = a1[r] + b1;
            cp[0]  = v0 > 0.f ? v0 : (__expf(v0) - 1.0f);
            cp[16] = v1 > 0.f ? v1 : (__expf(v1) - 1.0f);
            cp += Nd;
        }
    } else {
        #pragma unroll
        for (int r = 0; r < 8; ++r) {
            if (mBase + r < M) {
                float v0 = a0[r] + b0;
                float v1 = a1[r] + b1;
                cp[0]  = v0 > 0.f ? v0 : (__expf(v0) - 1.0f);
                cp[16] = v1 > 0.f ? v1 : (__expf(v1) - 1.0f);
            }
            cp += Nd;
        }
    }
}

__global__ __launch_bounds__(256, 1)
void gemm4_bias_elu(const float* __restrict__ A0, const float* __restrict__ A1,
                    const float* __restrict__ A2, const float* __restrict__ A3,
                    const float* __restrict__ B, const float* __restrict__ bias,
                    float* __restrict__ C, int M, int fin, int finLog2, int Nd) {
    __shared__ __bf16 lA[2][TM * ASTR];   // 2 * 10240 B
    __shared__ __bf16 lB[2][TN * BSTR];   // 2 *  5120 B  (transposed [n][k])

    const int Kd   = fin << 2;
    const int m0   = blockIdx.x * TM;
    const int n0   = blockIdx.y * TN;
    const int t    = threadIdx.x;
    const int wave = t >> 5;
    const int lane = t & 31;
    const int l16  = lane & 15;
    const int hi   = lane >> 4;
    const int msub = wave >> 1;
    const int nsub = wave & 1;

    // ---- per-thread loop-invariant addressing ----
    // A tile (128x32 f32): chunks t, t+256, t+512, t+768; row=ch>>3, col4=(ch&7)*4
    const int ch0 = t, ch1 = t + 256, ch2 = t + 512, ch3 = t + 768;
    int mA0 = m0 + (ch0 >> 3); if (mA0 >= M) mA0 = M - 1;
    int mA1 = m0 + (ch1 >> 3); if (mA1 >= M) mA1 = M - 1;
    int mA2 = m0 + (ch2 >> 3); if (mA2 >= M) mA2 = M - 1;
    int mA3 = m0 + (ch3 >> 3); if (mA3 >= M) mA3 = M - 1;
    const size_t aG0 = (size_t)mA0 * fin + (ch0 & 7) * 4;
    const size_t aG1 = (size_t)mA1 * fin + (ch1 & 7) * 4;
    const size_t aG2 = (size_t)mA2 * fin + (ch2 & 7) * 4;
    const size_t aG3 = (size_t)mA3 * fin + (ch3 & 7) * 4;
    const int aL0 = (ch0 >> 3) * ASTR + (ch0 & 7) * 4;
    const int aL1 = (ch1 >> 3) * ASTR + (ch1 & 7) * 4;
    const int aL2 = (ch2 >> 3) * ASTR + (ch2 & 7) * 4;
    const int aL3 = (ch3 >> 3) * ASTR + (ch3 & 7) * 4;
    // B tile (32x64 f32 -> LDS [n][k]): p = t, t+256; nn=p&63, k4=(p>>6)*4
    const int p0 = t, p1 = t + 256;
    const size_t bG0 = (size_t)((p0 >> 6) * 4) * Nd + n0 + (p0 & 63);
    const size_t bG1 = (size_t)((p1 >> 6) * 4) * Nd + n0 + (p1 & 63);
    const int bL0 = (p0 & 63) * BSTR + (p0 >> 6) * 4;
    const int bL1 = (p1 & 63) * BSTR + (p1 >> 6) * 4;
    // fragment LDS offsets (strip 0; strip 1 = +16 rows)
    const int aF = (msub * 32 + l16) * ASTR + hi * 8;
    const int bF = (nsub * 32 + l16) * BSTR + hi * 8;
    const int colBase = n0 + nsub * 32 + l16;

    // ---- staging registers (plain named vars -> SSA) ----
    v4f ra0, ra1, ra2, ra3;
    float rb00, rb01, rb02, rb03, rb10, rb11, rb12, rb13;

    // prologue: fetch + commit tile 0 (j=0, off=0)
    ra0 = *(const v4f*)(A0 + aG0);
    ra1 = *(const v4f*)(A0 + aG1);
    ra2 = *(const v4f*)(A0 + aG2);
    ra3 = *(const v4f*)(A0 + aG3);
    {
        const float* bp0 = B + bG0;
        const float* bp1 = B + bG1;
        rb00 = bp0[0]; rb01 = bp0[(size_t)Nd]; rb02 = bp0[(size_t)Nd * 2]; rb03 = bp0[(size_t)Nd * 3];
        rb10 = bp1[0]; rb11 = bp1[(size_t)Nd]; rb12 = bp1[(size_t)Nd * 2]; rb13 = bp1[(size_t)Nd * 3];
    }
    {
        __bf16* dA = &lA[0][0];
        __bf16* dB = &lB[0][0];
        *(v4bf*)(dA + aL0) = __builtin_convertvector(ra0, v4bf);
        *(v4bf*)(dA + aL1) = __builtin_convertvector(ra1, v4bf);
        *(v4bf*)(dA + aL2) = __builtin_convertvector(ra2, v4bf);
        *(v4bf*)(dA + aL3) = __builtin_convertvector(ra3, v4bf);
        v4f t0 = { rb00, rb01, rb02, rb03 };
        v4f t1 = { rb10, rb11, rb12, rb13 };
        *(v4bf*)(dB + bL0) = __builtin_convertvector(t0, v4bf);
        *(v4bf*)(dB + bL1) = __builtin_convertvector(t1, v4bf);
    }
    __syncthreads();

    v8f acc00 = {}, acc01 = {}, acc10 = {}, acc11 = {};
    int cur = 0;

    for (int k0 = 0; ; k0 += TK) {
        const int  kN = k0 + TK;
        const bool hasNext = kN < Kd;
        if (hasNext) {
            const int j   = kN >> finLog2;                  // uniform hop select
            const int off = kN & (fin - 1);
            const float* Aj = (j == 0) ? A0 : (j == 1) ? A1 : (j == 2) ? A2 : A3;
            ra0 = *(const v4f*)(Aj + aG0 + off);
            ra1 = *(const v4f*)(Aj + aG1 + off);
            ra2 = *(const v4f*)(Aj + aG2 + off);
            ra3 = *(const v4f*)(Aj + aG3 + off);
            const float* bp0 = B + bG0 + (size_t)kN * Nd;
            const float* bp1 = B + bG1 + (size_t)kN * Nd;
            rb00 = bp0[0]; rb01 = bp0[(size_t)Nd]; rb02 = bp0[(size_t)Nd * 2]; rb03 = bp0[(size_t)Nd * 3];
            rb10 = bp1[0]; rb11 = bp1[(size_t)Nd]; rb12 = bp1[(size_t)Nd * 2]; rb13 = bp1[(size_t)Nd * 3];
        }

        const __bf16* As = &lA[cur][0];
        const __bf16* Bs = &lB[cur][0];
        // A frag: lanes 0-15 row M=l16, K 0-7 & 16-23; lanes 16-31 K 8-15 & 24-31
        // B frag: lane column N=l16; lanes 0-15 K 0-7 & 16-23; lanes 16-31 K 8-15 & 24-31
        const v16bf af0 = cat16(*(const v8bf*)(As + aF),
                                *(const v8bf*)(As + aF + 16));
        const v16bf af1 = cat16(*(const v8bf*)(As + aF + 16 * ASTR),
                                *(const v8bf*)(As + aF + 16 * ASTR + 16));
        const v16bf bf0 = cat16(*(const v8bf*)(Bs + bF),
                                *(const v8bf*)(Bs + bF + 16));
        const v16bf bf1 = cat16(*(const v8bf*)(Bs + bF + 16 * BSTR),
                                *(const v8bf*)(Bs + bF + 16 * BSTR + 16));

        acc00 = __builtin_amdgcn_wmma_f32_16x16x32_bf16(false, af0, false, bf0, (short)0, acc00, false, false);
        acc01 = __builtin_amdgcn_wmma_f32_16x16x32_bf16(false, af0, false, bf1, (short)0, acc01, false, false);
        acc10 = __builtin_amdgcn_wmma_f32_16x16x32_bf16(false, af1, false, bf0, (short)0, acc10, false, false);
        acc11 = __builtin_amdgcn_wmma_f32_16x16x32_bf16(false, af1, false, bf1, (short)0, acc11, false, false);

        if (!hasNext) break;

        __bf16* dA = &lA[cur ^ 1][0];
        __bf16* dB = &lB[cur ^ 1][0];
        *(v4bf*)(dA + aL0) = __builtin_convertvector(ra0, v4bf);
        *(v4bf*)(dA + aL1) = __builtin_convertvector(ra1, v4bf);
        *(v4bf*)(dA + aL2) = __builtin_convertvector(ra2, v4bf);
        *(v4bf*)(dA + aL3) = __builtin_convertvector(ra3, v4bf);
        v4f t0 = { rb00, rb01, rb02, rb03 };
        v4f t1 = { rb10, rb11, rb12, rb13 };
        *(v4bf*)(dB + bL0) = __builtin_convertvector(t0, v4bf);
        *(v4bf*)(dB + bL1) = __builtin_convertvector(t1, v4bf);
        __syncthreads();
        cur ^= 1;
    }

    // fused epilogue: out = elu(acc + bias)
    const float b0 = bias[colBase];
    const float b1 = bias[colBase + 16];
    {
        const int mBase = m0 + msub * 32 + hi * 8;
        store_strip(C + (size_t)mBase * Nd + colBase, mBase, M, Nd, acc00, acc01, b0, b1);
    }
    {
        const int mBase = m0 + msub * 32 + 16 + hi * 8;
        store_strip(C + (size_t)mBase * Nd + colBase, mBase, M, Nd, acc10, acc11, b0, b1);
    }
}

// ------------------------------ classifier ---------------------------------
__global__ __launch_bounds__(128)
void classifier_kernel(const float* __restrict__ xw, const float* __restrict__ xn,
                       const float* __restrict__ Wc, const float* __restrict__ bc,
                       float* __restrict__ out, int H) {
    int n = blockIdx.x;
    __shared__ float s0[128];
    __shared__ float s1[128];
    float a0 = 0.f, a1 = 0.f;
    for (int h = threadIdx.x; h < H; h += 128) {
        float vw = xw[(size_t)n * H + h];
        float vn = xn[(size_t)n * H + h];
        a0 += vw * Wc[h * 2 + 0] + vn * Wc[(H + h) * 2 + 0];
        a1 += vw * Wc[h * 2 + 1] + vn * Wc[(H + h) * 2 + 1];
    }
    s0[threadIdx.x] = a0; s1[threadIdx.x] = a1;
    __syncthreads();
    for (int o = 64; o > 0; o >>= 1) {
        if (threadIdx.x < (unsigned)o) {
            s0[threadIdx.x] += s0[threadIdx.x + o];
            s1[threadIdx.x] += s1[threadIdx.x + o];
        }
        __syncthreads();
    }
    if (threadIdx.x == 0) {
        out[(size_t)n * 2 + 0] = s0[0] + bc[0];
        out[(size_t)n * 2 + 1] = s1[0] + bc[1];
    }
}

// ------------------------------ host orchestration -------------------------

extern "C" void kernel_launch(void* const* d_in, const int* in_sizes, int n_in,
                              void* d_out, int out_size, void* d_ws, size_t ws_size,
                              hipStream_t stream) {
    const int Fdim = 128, Hdim = 512;
    const float* x  = (const float*)d_in[0];
    const int*   ei = (const int*)d_in[1];
    const float* w  = (const float*)d_in[2];
    const int Nn = in_sizes[0] / Fdim;
    const int Ee = in_sizes[2];
    const int* src = ei;
    const int* dst = ei + Ee;

    // workspace layout (floats)
    float* ws     = (float*)d_ws;
    float* deg_w  = ws;
    float* deg_n  = ws + Nn;
    float* norm_w = ws + 2 * (size_t)Nn;
    float* norm_n = norm_w + Ee;
    float* base   = norm_n + Ee;
    const size_t NH = (size_t)Nn * Hdim;
    float* buf0 = base;
    float* buf1 = buf0 + NH;
    float* h1   = buf1 + NH;
    float* h2   = h1 + NH;
    float* h3   = h2 + NH;
    float* xw   = h3 + NH;
    float* xn   = xw + NH;

    auto zero_f = [&](float* p, size_t n) {
        zero_kernel<<<(unsigned)((n + 255) / 256), 256, 0, stream>>>(p, n);
    };

    // gcn_norm for both branches
    zero_f(deg_w, 2 * (size_t)Nn);
    deg_kernel<<<(Ee + 255) / 256, 256, 0, stream>>>(dst, w, deg_w, deg_n, Ee);
    norm_kernel<<<(Ee + 255) / 256, 256, 0, stream>>>(src, dst, w, deg_w, deg_n,
                                                      norm_w, norm_n, Ee);

    // one TAGConv layer: 3 SpMM hops then one fused GEMM over [x|h1|h2|h3]
    auto tag_layer = [&](const float* act_in, int fin, int finLog2,
                         const float* Wl, const float* bl,
                         const float* norm, float* out_act) {
        const float* hprev = act_in;
        float* hops[3] = { h1, h2, h3 };
        int F4 = fin / 4;
        long total = (long)Ee * F4;
        for (int k = 0; k < 3; ++k) {
            zero_f(hops[k], (size_t)Nn * fin);
            spmm_kernel<<<(unsigned)((total + 255) / 256), 256, 0, stream>>>(
                src, dst, norm, hprev, hops[k], Ee, F4);
            hprev = hops[k];
        }
        dim3 g((Nn + TM - 1) / TM, Hdim / TN);
        gemm4_bias_elu<<<g, 256, 0, stream>>>(act_in, h1, h2, h3, Wl, bl, out_act,
                                              Nn, fin, finLog2, Hdim);
    };

    const float* c1wW = (const float*)d_in[3];  const float* c1wb = (const float*)d_in[4];
    const float* c2wW = (const float*)d_in[5];  const float* c2wb = (const float*)d_in[6];
    const float* c3wW = (const float*)d_in[7];  const float* c3wb = (const float*)d_in[8];
    const float* c1nW = (const float*)d_in[9];  const float* c1nb = (const float*)d_in[10];
    const float* c2nW = (const float*)d_in[11]; const float* c2nb = (const float*)d_in[12];
    const float* c3nW = (const float*)d_in[13]; const float* c3nb = (const float*)d_in[14];
    const float* clsW = (const float*)d_in[15]; const float* clsb = (const float*)d_in[16];

    // weighted branch
    tag_layer(x,    Fdim, 7, c1wW, c1wb, norm_w, buf0);
    tag_layer(buf0, Hdim, 9, c2wW, c2wb, norm_w, buf1);
    tag_layer(buf1, Hdim, 9, c3wW, c3wb, norm_w, xw);
    // unit-weight branch
    tag_layer(x,    Fdim, 7, c1nW, c1nb, norm_n, buf0);
    tag_layer(buf0, Hdim, 9, c2nW, c2nb, norm_n, buf1);
    tag_layer(buf1, Hdim, 9, c3nW, c3nb, norm_n, xn);

    classifier_kernel<<<Nn, 128, 0, stream>>>(xw, xn, clsW, clsb, (float*)d_out, Hdim);
}
